// kpts_decoder_multistructure_43258910605498
// MI455X (gfx1250) — matrix operand order repaired and checked by reading, not compiled, not run on hardware
//
#include <hip/hip_runtime.h>
#include <cstddef>

// ---------------------------------------------------------------------------
// MI455X (gfx1250) implementation: bf16 WMMA GEMM pipeline, N-blocked and
// B-fragment software-pipelined.
//   L0:  h  = x @ w0 + b0            (M=64,   K=1024,  N=27392) -> hpA[64,128,256]
//   L1:  elu(spiral(hpA))            (M=4096/2752, K=67/46*256, N=256) -> hpB
//   L2:  elu(spiral(hpB))            (K=*256, N=128) -> hpA
//   L3:  elu(spiral(hpA))            (K=*128, N=64)  -> hpB
//   L4:  spiral(hpB)                 (K=*64,  N=2->pad 16) -> d_out[64,107,2]
// Weights pre-converted per call to transposed bf16 Wt[N][K] so each lane's
// B fragment is one contiguous 32B load. Gather fused into A fragment loads.
// Each wave computes a 16 x (16*NB) tile: A fragment converted once per
// K-step; all NB B fragments are loaded first (one clause, loads in flight),
// then NB WMMAs issue back-to-back.
// ---------------------------------------------------------------------------

typedef __attribute__((ext_vector_type(16))) __bf16 v16bf;
typedef __attribute__((ext_vector_type(8)))  float  v8f;

#define WAVES_PER_BLOCK 8

struct GemmParams {
  const float*  A;          // direct A (L0) or hp_in [B,128,CIN]
  const int*    idx;        // gather table [rows_per_b, S] (GATHER only)
  const __bf16* Wt;         // [Npad, K] transposed bf16 weights
  const float*  bias;       // [N_real]
  float*        out;
  int K;                    // Wt row length (= S*CIN for gather layers)
  int S;                    // spiral length (gather)
  int rows_per_b;           // rows of this GEMM per batch element
  int Mtiles;               // M / 16
  int N_real;               // valid output columns (bias length)
  int out_npb;              // node slots per batch in output buffer
  int node_base;            // node offset (0 inner, 64 outer)
  int ld_out;               // output channel stride
};

// A fragment for V_WMMA_F32_16X16X32_BF16 (16x32, row m = lane&15):
//   half = lane>>4: elements 0..15 -> k0 + half*16 + [0..16)
// rowp already points at (row, k0 + half*16).
__device__ __forceinline__ v16bf cvt_a_frag(const float* rowp) {
  float4 a0 = *(const float4*)(rowp);
  float4 a1 = *(const float4*)(rowp + 4);
  float4 a2 = *(const float4*)(rowp + 8);
  float4 a3 = *(const float4*)(rowp + 12);
  v16bf f;
  f[0]  = (__bf16)a0.x; f[1]  = (__bf16)a0.y; f[2]  = (__bf16)a0.z; f[3]  = (__bf16)a0.w;
  f[4]  = (__bf16)a1.x; f[5]  = (__bf16)a1.y; f[6]  = (__bf16)a1.z; f[7]  = (__bf16)a1.w;
  f[8]  = (__bf16)a2.x; f[9]  = (__bf16)a2.y; f[10] = (__bf16)a2.z; f[11] = (__bf16)a2.w;
  f[12] = (__bf16)a3.x; f[13] = (__bf16)a3.y; f[14] = (__bf16)a3.z; f[15] = (__bf16)a3.w;
  return f;
}

// One wave computes one 16(M) x 16*NB(N) tile. CIN==0 -> direct mode.
template<bool GATHER, bool ELU, int CIN, int NB>
__global__ __launch_bounds__(256) void gemm_wmma(GemmParams p) {
  const int lane = threadIdx.x & 31;
  const int wave = threadIdx.x >> 5;
  const int T    = blockIdx.x * WAVES_PER_BLOCK + wave;   // M tile index
  if (T >= p.Mtiles) return;                              // wave-uniform guard
  const int tileN = blockIdx.y;                           // N block of NB tiles
  const int half  = lane >> 4;
  const int l15   = lane & 15;

  const int r = T * 16 + l15;                             // A row for this lane
  int b_a = 0, n_a = 0;
  if (GATHER) { b_a = r / p.rows_per_b; n_a = r - b_a * p.rows_per_b; }

  // B fragment base: lane holds n = l15 (+16*nb), K-half = half.
  const __bf16* bbase =
      p.Wt + (size_t)(tileN * (16 * NB) + l15) * p.K + half * 16;
  const size_t bstride = (size_t)16 * p.K;                // next N tile

  v8f acc[NB] = {};

  if (GATHER) {
    const int*   idxrow = p.idx + (size_t)n_a * p.S;
    const float* abase  = p.A + (size_t)b_a * 128 * CIN + half * 16;
    for (int j = 0; j < p.S; ++j) {
      const int node = idxrow[j];                         // 0..127 (pad rows zeroed)
      const float*  rowp = abase + (size_t)node * CIN;
      const __bf16* bp   = bbase + (size_t)j * CIN;
      // Prefetch the next gathered A row while this K-group computes.
      const int nnode = (j + 1 < p.S) ? idxrow[j + 1] : node;
      __builtin_prefetch(abase + (size_t)nnode * CIN, 0, 1);
      __builtin_prefetch(bp + CIN, 0, 1);                 // next node's B slice
#pragma unroll
      for (int cb = 0; cb < CIN / 32; ++cb) {
        v16bf af = cvt_a_frag(rowp + cb * 32);
        v16bf bfr[NB];
#pragma unroll
        for (int nb = 0; nb < NB; ++nb)                   // burst all B loads
          bfr[nb] = *(const v16bf*)(bp + nb * bstride + cb * 32);
#pragma unroll
        for (int nb = 0; nb < NB; ++nb)                   // dense WMMA chain
          acc[nb] = __builtin_amdgcn_wmma_f32_16x16x32_bf16(
              false, af, false, bfr[nb], (short)0, acc[nb], false, false);
      }
    }
  } else {
    const float* rowp = p.A + (size_t)r * p.K + half * 16;
    const int nblk = p.K / 32;
    for (int kb = 0; kb < nblk; ++kb) {
      v16bf af = cvt_a_frag(rowp + (size_t)kb * 32);
      v16bf bfr[NB];
#pragma unroll
      for (int nb = 0; nb < NB; ++nb)
        bfr[nb] = *(const v16bf*)(bbase + nb * bstride + (size_t)kb * 32);
#pragma unroll
      for (int nb = 0; nb < NB; ++nb)
        acc[nb] = __builtin_amdgcn_wmma_f32_16x16x32_bf16(
            false, af, false, bfr[nb], (short)0, acc[nb], false, false);
    }
  }

  // D layout: lane holds column n = l15, rows m = half*8 + v.
#pragma unroll
  for (int nb = 0; nb < NB; ++nb) {
    const int col = (tileN * NB + nb) * 16 + l15;
    if (col < p.N_real) {
      const float bv = p.bias[col];
#pragma unroll
      for (int v = 0; v < 8; ++v) {
        const int m  = T * 16 + half * 8 + v;
        const int b  = m / p.rows_per_b;
        const int nn = m - b * p.rows_per_b;
        float val = acc[nb][v] + bv;
        if (ELU) val = val > 0.f ? val : (__expf(val) - 1.f);
        p.out[((size_t)b * p.out_npb + p.node_base + nn) * p.ld_out + col] = val;
      }
    }
  }
}

// W[K,N] f32 -> Wt[Npad,K] bf16 (transpose + convert; pad columns zeroed).
__global__ void prep_wt(const float* __restrict__ W, __bf16* __restrict__ Wt,
                        int K, int N, int Npad) {
  long long t = (long long)blockIdx.x * 256 + threadIdx.x;
  long long total = (long long)K * Npad;
  if (t >= total) return;
  int n = (int)(t % Npad);
  int k = (int)(t / Npad);
  Wt[(size_t)n * K + k] = (n < N) ? (__bf16)W[(size_t)k * N + n] : (__bf16)0.0f;
}

// Zero pad rows 107..127 of hp[B,128,c] so the next layer's gather reads zeros.
__global__ void zpad(float* hp, int c) {
  int t = blockIdx.x * 256 + threadIdx.x;
  int total = 64 * 21 * c;
  if (t >= total) return;
  int b    = t / (21 * c);
  int rem  = t - b * (21 * c);
  int node = 107 + rem / c;
  int ch   = rem - (rem / c) * c;
  hp[((size_t)b * 128 + node) * c + ch] = 0.f;
}

extern "C" void kernel_launch(void* const* d_in, const int* in_sizes, int n_in,
                              void* d_out, int out_size, void* d_ws, size_t ws_size,
                              hipStream_t stream) {
  (void)in_sizes; (void)n_in; (void)out_size; (void)ws_size;
  const float* x   = (const float*)d_in[0];
  const float* w0  = (const float*)d_in[1];
  const float* b0  = (const float*)d_in[2];
  const float* wi1 = (const float*)d_in[3];  const float* bi1 = (const float*)d_in[4];
  const float* wo1 = (const float*)d_in[5];  const float* bo1 = (const float*)d_in[6];
  const float* wi2 = (const float*)d_in[7];  const float* bi2 = (const float*)d_in[8];
  const float* wo2 = (const float*)d_in[9];  const float* bo2 = (const float*)d_in[10];
  const float* wi3 = (const float*)d_in[11]; const float* bi3 = (const float*)d_in[12];
  const float* wo3 = (const float*)d_in[13]; const float* bo3 = (const float*)d_in[14];
  const float* wi4 = (const float*)d_in[15]; const float* bi4 = (const float*)d_in[16];
  const float* wo4 = (const float*)d_in[17]; const float* bo4 = (const float*)d_in[18];
  const int* idx_inner = (const int*)d_in[19];
  const int* idx_outer = (const int*)d_in[20];
  float* out = (float*)d_out;

  // Workspace carve-out (256B aligned slices), ~97 MB total.
  char* ws = (char*)d_ws;
  size_t off = 0;
  auto carve = [&](size_t bytes) -> void* {
    void* p = ws + off;
    off += (bytes + 255) & ~(size_t)255;
    return p;
  };
  float*  hpA  = (float*)carve(64ull * 128 * 256 * 4);
  float*  hpB  = (float*)carve(64ull * 128 * 256 * 4);
  __bf16* w0t  = (__bf16*)carve(27392ull * 1024 * 2);
  __bf16* wi1t = (__bf16*)carve(256ull * 17152 * 2);
  __bf16* wo1t = (__bf16*)carve(256ull * 11776 * 2);
  __bf16* wi2t = (__bf16*)carve(128ull * 17152 * 2);
  __bf16* wo2t = (__bf16*)carve(128ull * 11776 * 2);
  __bf16* wi3t = (__bf16*)carve(64ull * 8576 * 2);
  __bf16* wo3t = (__bf16*)carve(64ull * 5888 * 2);
  __bf16* wi4t = (__bf16*)carve(16ull * 4288 * 2);
  __bf16* wo4t = (__bf16*)carve(16ull * 2944 * 2);

  auto prep = [&](const float* W, __bf16* Wt, int K, int N, int Npad) {
    long long total = (long long)K * Npad;
    int blocks = (int)((total + 255) / 256);
    prep_wt<<<blocks, 256, 0, stream>>>(W, Wt, K, N, Npad);
  };
  prep(w0,  w0t,  1024,  27392, 27392);
  prep(wi1, wi1t, 17152, 256,   256);
  prep(wo1, wo1t, 11776, 256,   256);
  prep(wi2, wi2t, 17152, 128,   128);
  prep(wo2, wo2t, 11776, 128,   128);
  prep(wi3, wi3t, 8576,  64,    64);
  prep(wo3, wo3t, 5888,  64,    64);
  prep(wi4, wi4t, 4288,  2,     16);
  prep(wo4, wo4t, 2944,  2,     16);

  // ---- Layer 0: x @ w0 + b0 -> hpA [64,128,256] (node folded into col) ----
  {
    GemmParams p{};
    p.A = x; p.idx = nullptr; p.Wt = w0t; p.bias = b0; p.out = hpA;
    p.K = 1024; p.S = 0; p.rows_per_b = 1; p.Mtiles = 4;
    p.N_real = 27392; p.out_npb = 128; p.node_base = 0; p.ld_out = 256;
    dim3 g(1, 27392 / 64);   // Mtiles=4 -> 1 block.x; NB=4 -> 64 cols per block.y
    gemm_wmma<false, false, 0, 4><<<g, 256, 0, stream>>>(p);
  }
  zpad<<<(64 * 21 * 256 + 255) / 256, 256, 0, stream>>>(hpA, 256);

  // Spiral region descriptors.
  struct Reg { const int* idx; int S; int rpb; int Mtiles; int node_base; };
  const Reg RIN  = { idx_inner, 67, 64, 4096 / 16, 0 };
  const Reg ROUT = { idx_outer, 46, 43, 2752 / 16, 64 };

  auto fill = [&](const float* hin, float* hout,
                  const __bf16* Wt, const float* bias,
                  int cin, int cout, const Reg& rg, bool final_out) {
    GemmParams p{};
    p.A = hin; p.idx = rg.idx; p.Wt = Wt; p.bias = bias; p.out = hout;
    p.K = rg.S * cin; p.S = rg.S; p.rows_per_b = rg.rpb; p.Mtiles = rg.Mtiles;
    p.N_real = cout;
    p.out_npb = final_out ? 107 : 128;
    p.node_base = rg.node_base;
    p.ld_out = final_out ? 2 : cout;
    return p;
  };

  // ---- Layer 1: 256 -> 256, ELU, hpA -> hpB ----
  {
    GemmParams p = fill(hpA, hpB, wi1t, bi1, 256, 256, RIN, false);
    dim3 g((p.Mtiles + 7) / 8, 256 / 64);
    gemm_wmma<true, true, 256, 4><<<g, 256, 0, stream>>>(p);
    p = fill(hpA, hpB, wo1t, bo1, 256, 256, ROUT, false);
    dim3 g2((p.Mtiles + 7) / 8, 256 / 64);
    gemm_wmma<true, true, 256, 4><<<g2, 256, 0, stream>>>(p);
  }
  zpad<<<(64 * 21 * 256 + 255) / 256, 256, 0, stream>>>(hpB, 256);

  // ---- Layer 2: 256 -> 128, ELU, hpB -> hpA ----
  {
    GemmParams p = fill(hpB, hpA, wi2t, bi2, 256, 128, RIN, false);
    dim3 g((p.Mtiles + 7) / 8, 128 / 64);
    gemm_wmma<true, true, 256, 4><<<g, 256, 0, stream>>>(p);
    p = fill(hpB, hpA, wo2t, bo2, 256, 128, ROUT, false);
    dim3 g2((p.Mtiles + 7) / 8, 128 / 64);
    gemm_wmma<true, true, 256, 4><<<g2, 256, 0, stream>>>(p);
  }
  zpad<<<(64 * 21 * 128 + 255) / 256, 256, 0, stream>>>(hpA, 128);

  // ---- Layer 3: 128 -> 64, ELU, hpA -> hpB ----
  {
    GemmParams p = fill(hpA, hpB, wi3t, bi3, 128, 64, RIN, false);
    dim3 g((p.Mtiles + 7) / 8, 1);
    gemm_wmma<true, true, 128, 4><<<g, 256, 0, stream>>>(p);
    p = fill(hpA, hpB, wo3t, bo3, 128, 64, ROUT, false);
    dim3 g2((p.Mtiles + 7) / 8, 1);
    gemm_wmma<true, true, 128, 4><<<g2, 256, 0, stream>>>(p);
  }
  zpad<<<(64 * 21 * 64 + 255) / 256, 256, 0, stream>>>(hpB, 64);

  // ---- Layer 4: 64 -> 2 (N padded to 16), no ELU, hpB -> d_out [64,107,2] ----
  {
    GemmParams p = fill(hpB, out, wi4t, bi4, 64, 2, RIN, true);
    dim3 g((p.Mtiles + 7) / 8, 1);
    gemm_wmma<true, false, 64, 1><<<g, 256, 0, stream>>>(p);
    p = fill(hpB, out, wo4t, bo4, 64, 2, ROUT, true);
    dim3 g2((p.Mtiles + 7) / 8, 1);
    gemm_wmma<true, false, 64, 1><<<g2, 256, 0, stream>>>(p);
  }
}
